// MHA_84877143704222
// MI455X (gfx1250) — matrix operand order repaired and checked by reading, not compiled, not run on hardware
//
#include <hip/hip_runtime.h>
#include <hip/hip_bf16.h>
#include <math.h>
#include <stdint.h>

// ---------------------------------------------------------------------------
// MHA layer for MI455X (gfx1250, wave32). Pipeline:
//   k0: fp32 -> bf16 conversion of H / [Wq;Wk;Wv] / Wo   (bandwidth, ~5us)
//   k1: fused QKV GEMM, TDM tensor_load_to_lds double-buffered, WMMA bf16
//   k2: flash attention: TDM-staged K/V tiles, WMMA QK^T/PV, DPP row-max,
//       row-sums folded into a WMMA against an all-ones B fragment
//   k3: O projection GEMM (same TDM+WMMA pipeline), fp32 out
// ---------------------------------------------------------------------------

#ifndef __has_builtin
#define __has_builtin(x) 0
#endif
#if __has_builtin(__builtin_amdgcn_tensor_load_to_lds) && \
    __has_builtin(__builtin_amdgcn_s_wait_tensorcnt)
#define HAVE_TDM 1
#else
#define HAVE_TDM 0
#endif
#if __has_builtin(__builtin_amdgcn_update_dpp)
#define HAVE_DPP 1
#else
#define HAVE_DPP 0
#endif

typedef __attribute__((ext_vector_type(16))) __bf16       bf16x16;
typedef __attribute__((ext_vector_type(8)))  __bf16       bf16x8;
typedef __attribute__((ext_vector_type(8)))  float        f32x8;
typedef __attribute__((ext_vector_type(4)))  unsigned int u32x4;
typedef __attribute__((ext_vector_type(8)))  int          i32x8;
typedef __attribute__((ext_vector_type(4)))  int          i32x4;

#define B_SZ 2
#define SEQ  2048
#define DM   2048
#define NQ   32
#define NKV  8
#define HD   64
#define MTOT (B_SZ*SEQ)          // 4096
#define NQKV (DM + 2*NKV*HD)     // 3072
#define LDT  40                  // GEMM LDS tile row (elems): 80B, bank-friendly
#define PLD  72                  // P-tile row (elems): 144B
#define LDK  72                  // K/V LDS tile row (elems): 144B

__device__ __forceinline__ f32x8 wmma_bf16(bf16x16 a, bf16x16 b, f32x8 c) {
  return __builtin_amdgcn_wmma_f32_16x16x32_bf16(false, a, false, b, (short)0, c,
                                                 false, false);
}

__device__ __forceinline__ bf16x16 frag_cat(bf16x8 lo, bf16x8 hi) {
  return __builtin_shufflevector(lo, hi, 0,1,2,3,4,5,6,7,8,9,10,11,12,13,14,15);
}

// A fragment: 16(M) x 32(K) bf16, row-major, row stride ld (elems).
// ISA 7.12.2: lanes 0-15 -> row M=lane, K {0..7,16..23}; lanes 16-31 -> K {8..15,24..31}
__device__ __forceinline__ bf16x16 load_frag_a(const __bf16* p, int ld, int lane) {
  const int half = lane >> 4, r = lane & 15;
  const __bf16* q = p + r * ld + half * 8;
  return frag_cat(*(const bf16x8*)q, *(const bf16x8*)(q + 16));
}

// B fragment: 32(K) x 16(N) bf16, stored transposed (N rows, K contiguous), stride ld.
__device__ __forceinline__ bf16x16 load_frag_b(const __bf16* p, int ld, int lane) {
  const int half = lane >> 4, n = lane & 15;
  const __bf16* q = p + n * ld + half * 8;
  return frag_cat(*(const bf16x8*)q, *(const bf16x8*)(q + 16));
}

// 16-lane max reduction, pure VALU via DPP16 (no DS pipe, no waits):
// quad_perm xor1, quad_perm xor2, row_ror:4, row_ror:8.
#if HAVE_DPP
#define DPPMAX(x, c)                                                         \
  do {                                                                       \
    int _u = (int)__float_as_uint(x);                                        \
    int _y = __builtin_amdgcn_update_dpp(_u, _u, (c), 0xF, 0xF, false);      \
    (x) = fmaxf((x), __uint_as_float((unsigned)_y));                         \
  } while (0)
__device__ __forceinline__ float rowmax16(float x) {
  DPPMAX(x, 0xB1);   // quad_perm [1,0,3,2]
  DPPMAX(x, 0x4E);   // quad_perm [2,3,0,1]
  DPPMAX(x, 0x124);  // row_ror:4
  DPPMAX(x, 0x128);  // row_ror:8
  return x;
}
#else
__device__ __forceinline__ float rowmax16(float x) {
  x = fmaxf(x, __shfl_xor(x, 1));
  x = fmaxf(x, __shfl_xor(x, 2));
  x = fmaxf(x, __shfl_xor(x, 4));
  x = fmaxf(x, __shfl_xor(x, 8));
  return x;
}
#endif

#if HAVE_TDM
// TDM: DMA a ROWS x TW(bf16) 2D tile, global -> LDS, padding each row of
// TW*2 bytes with 16B in LDS. PADIV: pad_interval code (3 -> 64B, 4 -> 128B).
template <int TW, int ROWS, int PADIV>
__device__ __forceinline__ void tdm_load_tile(const __bf16* gsrc, void* ldsdst,
                                              int gstride_elems) {
  unsigned long long ga = (unsigned long long)(uintptr_t)gsrc;
  unsigned int lv = (unsigned int)(uintptr_t)ldsdst;   // low 32b = LDS byte addr
  u32x4 g0;
  g0[0] = 1u;                                          // count=1, user descriptor
  g0[1] = lv;                                          // lds_addr
  g0[2] = (unsigned int)ga;                            // global_addr[31:0]
  g0[3] = (unsigned int)((ga >> 32) & 0x01ffffffu) | (2u << 30);  // [56:32], type=2
  const unsigned int td0 = 1u << 24, td1 = 1u << 24;   // huge tensor dims (no OOB)
  i32x8 g1;
  g1[0] = (int)((1u << 16)                 // data_size = 1 -> 2 bytes
        |       (1u << 20)                 // pad_enable
        |       ((unsigned)PADIV << 22)    // pad_interval code
        |       (3u << 25));               // pad_amount code 3 -> 4 DWORDs (16B)
  g1[1] = (int)((td0 & 0xffffu) << 16);                  // tensor_dim0[15:0]
  g1[2] = (int)((td0 >> 16) | ((td1 & 0xffffu) << 16));  // dim0[31:16], dim1[15:0]
  g1[3] = (int)((td1 >> 16) | ((unsigned)TW << 16));     // dim1[31:16], tile_dim0
  g1[4] = (int)ROWS;                                     // tile_dim1, tile_dim2=0
  g1[5] = (int)(unsigned)gstride_elems;                  // tensor_dim0_stride[31:0]
  g1[6] = 0;
  g1[7] = 0;
  i32x4 z4 = {0, 0, 0, 0};
#if __clang_major__ >= 23
  i32x8 z8 = {0, 0, 0, 0, 0, 0, 0, 0};
  __builtin_amdgcn_tensor_load_to_lds(g0, g1, z4, z4, z8, 0);
#else
  __builtin_amdgcn_tensor_load_to_lds(g0, g1, z4, z4, 0);
#endif
}
#endif

// ------------------------------- kernel 0 ----------------------------------
__global__ __launch_bounds__(256) void cvt_kernel(const float* __restrict__ src,
                                                  __bf16* __restrict__ dst, int n) {
  int i = (blockIdx.x * 256 + threadIdx.x) * 8;
  if (i + 8 <= n) {
    float4 a = *(const float4*)(src + i);
    float4 b = *(const float4*)(src + i + 4);
    bf16x8 o;
    o[0] = (__bf16)a.x; o[1] = (__bf16)a.y; o[2] = (__bf16)a.z; o[3] = (__bf16)a.w;
    o[4] = (__bf16)b.x; o[5] = (__bf16)b.y; o[6] = (__bf16)b.z; o[7] = (__bf16)b.w;
    *(bf16x8*)(dst + i) = o;
  }
}

// ------------------------ shared GEMM compute step -------------------------
__device__ __forceinline__ void mm_step(const __bf16 (&tA)[128][LDT],
                                        const __bf16 (&tB)[128][LDT],
                                        int wm, int wn, int lane,
                                        f32x8 (&acc)[2][4]) {
  bf16x16 a0 = load_frag_a(&tA[wm*32     ][0], LDT, lane);
  bf16x16 a1 = load_frag_a(&tA[wm*32 + 16][0], LDT, lane);
  bf16x16 b0 = load_frag_b(&tB[wn*64     ][0], LDT, lane);
  bf16x16 b1 = load_frag_b(&tB[wn*64 + 16][0], LDT, lane);
  bf16x16 b2 = load_frag_b(&tB[wn*64 + 32][0], LDT, lane);
  bf16x16 b3 = load_frag_b(&tB[wn*64 + 48][0], LDT, lane);
  acc[0][0] = wmma_bf16(a0, b0, acc[0][0]);
  acc[0][1] = wmma_bf16(a0, b1, acc[0][1]);
  acc[0][2] = wmma_bf16(a0, b2, acc[0][2]);
  acc[0][3] = wmma_bf16(a0, b3, acc[0][3]);
  acc[1][0] = wmma_bf16(a1, b0, acc[1][0]);
  acc[1][1] = wmma_bf16(a1, b1, acc[1][1]);
  acc[1][2] = wmma_bf16(a1, b2, acc[1][2]);
  acc[1][3] = wmma_bf16(a1, b3, acc[1][3]);
}

// -------------------------- kernels 1 & 3 (GEMM) ---------------------------
// EPI==0: QKV epilogue (bf16 Q row-major, K l-major, V transposed)
// EPI==1: plain fp32 store to Co
template<int EPI>
__global__ __launch_bounds__(256) void gemm_kernel(
    const __bf16* __restrict__ A, const __bf16* __restrict__ Bt,
    __bf16* __restrict__ Qbf, __bf16* __restrict__ Kbf, __bf16* __restrict__ Vt,
    float* __restrict__ Co)
{
  __shared__ __align__(16) __bf16 shA[2][128][LDT];
  __shared__ __align__(16) __bf16 shB[2][128][LDT];
  const int tid = threadIdx.x, lane = tid & 31, w = tid >> 5;
  const int wm = w >> 1, wn = w & 1;
  const int m0 = blockIdx.y * 128, n0 = blockIdx.x * 128;
  f32x8 acc[2][4] = {};

#if HAVE_TDM
  // Double-buffered: wave0 DMAs the A tile, wave1 the B tile; one barrier/iter.
  auto issue = [&](int buf, int kb) {
    if (w == 0)
      tdm_load_tile<32, 128, 3>(A + (size_t)m0 * DM + kb, &shA[buf][0][0], DM);
    else if (w == 1)
      tdm_load_tile<32, 128, 3>(Bt + (size_t)n0 * DM + kb, &shB[buf][0][0], DM);
  };
  issue(0, 0);
  for (int kb = 0; kb < DM; kb += 32) {
    const int cur = (kb >> 5) & 1;
    __builtin_amdgcn_s_wait_tensorcnt(0);
    __syncthreads();
    if (kb + 32 < DM) issue(cur ^ 1, kb + 32);
    mm_step(shA[cur], shB[cur], wm, wn, lane, acc);
  }
#else
  // Fallback: cooperative batched b128 copies, single buffer.
  for (int kb = 0; kb < DM; kb += 32) {
    __syncthreads();
    {
      int row = tid >> 1, c = (tid & 1) * 16;
      bf16x8 a0 = *(const bf16x8*)(A  + (size_t)(m0 + row) * DM + kb + c);
      bf16x8 a1 = *(const bf16x8*)(A  + (size_t)(m0 + row) * DM + kb + c + 8);
      bf16x8 b0 = *(const bf16x8*)(Bt + (size_t)(n0 + row) * DM + kb + c);
      bf16x8 b1 = *(const bf16x8*)(Bt + (size_t)(n0 + row) * DM + kb + c + 8);
      *(bf16x8*)&shA[0][row][c]     = a0;
      *(bf16x8*)&shA[0][row][c + 8] = a1;
      *(bf16x8*)&shB[0][row][c]     = b0;
      *(bf16x8*)&shB[0][row][c + 8] = b1;
    }
    __syncthreads();
    mm_step(shA[0], shB[0], wm, wn, lane, acc);
  }
#endif

  // C layout (ISA 7.12.2): VGPR r, lane L -> row r + 8*(L>=16), col L%16
  const int col = lane & 15, half = lane >> 4;
#pragma unroll
  for (int mi = 0; mi < 2; ++mi)
#pragma unroll
    for (int ni = 0; ni < 4; ++ni) {
      int n = n0 + wn * 64 + ni * 16 + col;
#pragma unroll
      for (int r = 0; r < 8; ++r) {
        int m = m0 + wm * 32 + mi * 16 + r + 8 * half;
        float v = acc[mi][ni][r];
        if constexpr (EPI == 0) {
          if (n < DM) {
            Qbf[(size_t)m * DM + n] = (__bf16)v;
          } else if (n < DM + NKV * HD) {
            int nk = n - DM, kvh = nk >> 6, h = nk & 63;
            int b = m >> 11, s = m & (SEQ - 1);
            Kbf[(((size_t)(b * NKV + kvh)) * SEQ + s) * HD + h] = (__bf16)v;
          } else {
            int nv = n - DM - NKV * HD, kvh = nv >> 6, h = nv & 63;
            int b = m >> 11, s = m & (SEQ - 1);
            Vt[(((size_t)(b * NKV + kvh)) * HD + h) * SEQ + s] = (__bf16)v;
          }
        } else {
          Co[(size_t)m * DM + n] = v;
        }
      }
    }
}

// ------------------------------- kernel 2 ----------------------------------
__global__ __launch_bounds__(256) void flash_kernel(
    const __bf16* __restrict__ Qbf, const __bf16* __restrict__ Kbf,
    const __bf16* __restrict__ Vt,  __bf16* __restrict__ Obf)
{
  __shared__ __align__(16) __bf16 psh[8][16][PLD];   // per-wave P tile
  __shared__ __align__(16) __bf16 Ksh[2][64][LDK];   // K tile: [key][hd]
  __shared__ __align__(16) __bf16 Vsh[2][64][LDK];   // V tile: [hd][key]
  const int tid = threadIdx.x, lane = tid & 31, w = tid >> 5;
  const int half = lane >> 4, col = lane & 15;
  const int bx = blockIdx.x;
  const int mt = bx & 15, qh = (bx >> 4) & 31, b = bx >> 9;
  const int kvh = qh >> 2;                          // GQA group of 4
  const int q0 = mt * 128 + w * 16;

  const __bf16* Qp = Qbf + ((size_t)(b * SEQ + q0)) * DM + qh * HD;
  const __bf16* Kp = Kbf + ((size_t)(b * NKV + kvh)) * SEQ * HD;
  const __bf16* Vp = Vt  + ((size_t)(b * NKV + kvh)) * HD * SEQ;

  // Q fragments, pre-scaled by 1/sqrt(64) = 2^-3 (exact in bf16)
  bf16x16 qa0 = load_frag_a(Qp, DM, lane);       // head_dim 0..31
  bf16x16 qa1 = load_frag_a(Qp + 32, DM, lane);  // head_dim 32..63
#pragma unroll
  for (int i = 0; i < 16; ++i) {
    qa0[i] = (__bf16)((float)qa0[i] * 0.125f);
    qa1[i] = (__bf16)((float)qa1[i] * 0.125f);
  }
  // all-ones B fragment: P x ones accumulates the softmax row sums
  bf16x16 onesb;
#pragma unroll
  for (int i = 0; i < 16; ++i) onesb[i] = (__bf16)1.0f;

  float mrow[8];
  f32x8 acc[4] = {};
  f32x8 lacc = {};
#pragma unroll
  for (int r = 0; r < 8; ++r) mrow[r] = -3.0e38f;

#if HAVE_TDM
  // All 8 waves share (b, kvh): stage K/V tiles once per block via TDM,
  // double-buffered; wave0 DMAs K, wave1 DMAs V. One barrier per key tile.
  auto issue = [&](int buf, int kt) {
    if (w == 0)
      tdm_load_tile<64, 64, 4>(Kp + (size_t)kt * HD, &Ksh[buf][0][0], HD);
    else if (w == 1)
      tdm_load_tile<64, 64, 4>(Vp + kt, &Vsh[buf][0][0], SEQ);
  };
  issue(0, 0);
#endif

  for (int kt = 0; kt < SEQ; kt += 64) {
#if HAVE_TDM
    const int cur = (kt >> 6) & 1;
    __builtin_amdgcn_s_wait_tensorcnt(0);
    __syncthreads();
    if (kt + 64 < SEQ) issue(cur ^ 1, kt + 64);
#else
    const int cur = 0;
    __syncthreads();
    {
      // cooperative staging: 64 rows x 128B per tile; 256 threads x 32B
      int row = tid >> 2, seg = (tid & 3) * 16;
      const __bf16* kg = Kp + (size_t)(kt + row) * HD + seg;
      const __bf16* vg = Vp + (size_t)row * SEQ + kt + seg;
      bf16x8 k0 = *(const bf16x8*)kg, k1 = *(const bf16x8*)(kg + 8);
      bf16x8 v0 = *(const bf16x8*)vg, v1 = *(const bf16x8*)(vg + 8);
      *(bf16x8*)&Ksh[0][row][seg]     = k0;
      *(bf16x8*)&Ksh[0][row][seg + 8] = k1;
      *(bf16x8*)&Vsh[0][row][seg]     = v0;
      *(bf16x8*)&Vsh[0][row][seg + 8] = v1;
    }
    __syncthreads();
#endif

    // S = (Q/sqrt(d)) K^T  -> 16 x 64 scores (4 C tiles)
    f32x8 sc[4] = {};
#pragma unroll
    for (int t = 0; t < 4; ++t) {
      sc[t] = wmma_bf16(qa0, load_frag_b(&Ksh[cur][t*16][0],  LDK, lane), sc[t]);
      sc[t] = wmma_bf16(qa1, load_frag_b(&Ksh[cur][t*16][32], LDK, lane), sc[t]);
    }

    // row max: per-lane partials, then DPP 16-lane reduction (pure VALU)
    float so[8];
#pragma unroll
    for (int r = 0; r < 8; ++r) {
      float v = fmaxf(fmaxf(sc[0][r], sc[1][r]), fmaxf(sc[2][r], sc[3][r]));
      v = rowmax16(v);
      float mn = fmaxf(mrow[r], v);
      so[r] = __expf(mrow[r] - mn);
      mrow[r] = mn;
    }
    // P = exp(S - m_new); rescale running accumulators
#pragma unroll
    for (int t = 0; t < 4; ++t)
#pragma unroll
      for (int r = 0; r < 8; ++r)
        sc[t][r] = __expf(sc[t][r] - mrow[r]);
#pragma unroll
    for (int r = 0; r < 8; ++r) lacc[r] *= so[r];
#pragma unroll
    for (int t = 0; t < 4; ++t)
#pragma unroll
      for (int r = 0; r < 8; ++r)
        acc[t][r] *= so[r];

    // P (C layout) -> per-wave LDS, re-read as A fragments (same-wave, in-order)
#pragma unroll
    for (int t = 0; t < 4; ++t)
#pragma unroll
      for (int r = 0; r < 8; ++r)
        psh[w][r + 8 * half][t * 16 + col] = (__bf16)sc[t][r];
    asm volatile("s_wait_dscnt 0" ::: "memory");

    // O += P V and l += P 1 : keys in 2 chunks of 32, head_dim in 4 chunks of 16
#pragma unroll
    for (int kc = 0; kc < 2; ++kc) {
      bf16x16 pa = load_frag_a(&psh[w][0][kc * 32], PLD, lane);
      lacc = wmma_bf16(pa, onesb, lacc);
#pragma unroll
      for (int t = 0; t < 4; ++t)
        acc[t] = wmma_bf16(pa, load_frag_b(&Vsh[cur][t*16][kc*32], LDK, lane),
                           acc[t]);
    }
  }

  // normalize (one reciprocal per row) and store attention output
  float inv[8];
#pragma unroll
  for (int r = 0; r < 8; ++r) inv[r] = 1.0f / lacc[r];
#pragma unroll
  for (int t = 0; t < 4; ++t)
#pragma unroll
    for (int r = 0; r < 8; ++r) {
      float v = acc[t][r] * inv[r];
      int m = b * SEQ + q0 + r + 8 * half;
      int n = qh * HD + t * 16 + col;
      Obf[(size_t)m * DM + n] = (__bf16)v;
    }
}

// ------------------------------- launcher ----------------------------------
extern "C" void kernel_launch(void* const* d_in, const int* in_sizes, int n_in,
                              void* d_out, int out_size, void* d_ws, size_t ws_size,
                              hipStream_t stream) {
  const float* H  = (const float*)d_in[0];
  const float* Wq = (const float*)d_in[1];
  const float* Wk = (const float*)d_in[2];
  const float* Wv = (const float*)d_in[3];
  const float* Wo = (const float*)d_in[4];
  float* out = (float*)d_out;

  // workspace layout (bf16 elems):
  //  Hbf 8M | Wcat 6M | Wobf 4M | Q 8M | K 2M | V^T 2M | attn-out 8M  (76MB)
  char* ws = (char*)d_ws;
  const size_t eH = (size_t)MTOT * DM;          // 8M
  const size_t eWc = (size_t)NQKV * DM;         // 6M
  const size_t eWo = (size_t)DM * DM;           // 4M
  const size_t eKV = (size_t)B_SZ * NKV * SEQ * HD;  // 2M
  __bf16* Hbf  = (__bf16*)(ws);
  __bf16* Wcat = Hbf + eH;
  __bf16* Wobf = Wcat + eWc;
  __bf16* Qbf  = Wobf + eWo;
  __bf16* Kbf  = Qbf + eH;
  __bf16* Vt   = Kbf + eKV;
  __bf16* Obf  = Vt + eKV;

  // k0: convert everything to bf16 (weights concatenated: [Wq;Wk;Wv] rows)
  cvt_kernel<<<dim3((int)(eH / 2048)), 256, 0, stream>>>(H, Hbf, (int)eH);
  cvt_kernel<<<dim3(2048), 256, 0, stream>>>(Wq, Wcat, DM * DM);
  cvt_kernel<<<dim3(512),  256, 0, stream>>>(Wk, Wcat + (size_t)DM * DM, NKV * HD * DM);
  cvt_kernel<<<dim3(512),  256, 0, stream>>>(Wv, Wcat + (size_t)(DM + NKV * HD) * DM,
                                             NKV * HD * DM);
  cvt_kernel<<<dim3(2048), 256, 0, stream>>>(Wo, Wobf, DM * DM);

  // k1: fused QKV projection
  gemm_kernel<0><<<dim3(NQKV / 128, MTOT / 128), 256, 0, stream>>>(
      Hbf, Wcat, Qbf, Kbf, Vt, nullptr);
  // k2: attention
  flash_kernel<<<dim3(B_SZ * NQ * (SEQ / 128)), 256, 0, stream>>>(
      Qbf, Kbf, Vt, Obf);
  // k3: O projection
  gemm_kernel<1><<<dim3(DM / 128, MTOT / 128), 256, 0, stream>>>(
      Obf, Wobf, nullptr, nullptr, nullptr, out);
}
